// LlamaQAttention2_21947282883011
// MI455X (gfx1250) — matrix operand order repaired and checked by reading, not compiled, not run on hardware
//
#include <hip/hip_runtime.h>
#include <math.h>

// ---------------------------------------------------------------------------
// LlamaQAttention (AWQ int4 QKV/O proj + RoPE + causal attention) for gfx1250
// Matmuls on v_wmma_f32_16x16x32_f16; GEMM A-tiles staged with the CDNA5
// async-to-LDS path (global_load_async_to_lds_b128 + s_wait_asynccnt).
// ---------------------------------------------------------------------------

typedef __attribute__((ext_vector_type(16))) _Float16 v16h;
typedef __attribute__((ext_vector_type(8)))  _Float16 v8h;
typedef __attribute__((ext_vector_type(8)))  float    v8f;

union Frag16 { v16h v; v8h h[2]; };
union H8     { v8h v; _Float16 e[8]; };

__device__ __forceinline__ v8f wmma_f16(const Frag16& a, const Frag16& b, v8f c) {
  return __builtin_amdgcn_wmma_f32_16x16x32_f16(
      /*neg_a=*/false, a.v, /*neg_b=*/false, b.v,
      /*c_mod=*/(short)0, c, /*reuse_a=*/false, /*reuse_b=*/false);
}

#define BATCH   2
#define SEQ     1024
#define HDIM    4096
#define NHEADS  32
#define DHEAD   128
#define QKV_N   (3 * HDIM)   // 12288

// ---------------------------------------------------------------------------
// AWQ int4 dequant:  w = (nibble(q) - nibble(z)) * scale, group size 128,
// nibble order AWQ_ORDER = [0,4,1,5,2,6,3,7]  (shifts 0,16,4,20,8,24,12,28)
// ---------------------------------------------------------------------------
__global__ void dequant_awq(const int* __restrict__ qw, const int* __restrict__ qz,
                            const float* __restrict__ sc, _Float16* __restrict__ W,
                            int K, int N) {
  const int NP = N >> 3;
  int idx = blockIdx.x * blockDim.x + threadIdx.x;
  if (idx >= K * NP) return;
  int pc = idx % NP;
  int k  = idx / NP;
  int g  = k >> 7;  // group of 128 along in-dim
  unsigned int w = (unsigned int)qw[idx];
  unsigned int z = (unsigned int)qz[g * NP + pc];
  const int shifts[8] = {0, 16, 4, 20, 8, 24, 12, 28};
  const float* sp = sc + (size_t)g * N + pc * 8;
  H8 o;
#pragma unroll
  for (int i = 0; i < 8; ++i) {
    int wn = (w >> shifts[i]) & 0xF;
    int zn = (z >> shifts[i]) & 0xF;
    o.e[i] = (_Float16)((float)(wn - zn) * sp[i]);
  }
  *(v8h*)(W + (size_t)k * N + pc * 8) = o.v;   // 16B store
}

__global__ void f32_to_f16(const float* __restrict__ x, _Float16* __restrict__ y, int n) {
  int i = blockIdx.x * blockDim.x + threadIdx.x;
  if (i < n) y[i] = (_Float16)x[i];
}

// ---------------------------------------------------------------------------
// WMMA GEMM:  C[M,N] = A[M,K] * B[K,N]   (A,B f16 row-major, f32 accum)
// Block 256 thr = 8 waves, tile 128x256, K-step 32.
// Wave (wm,wn) in 2x4 grid owns 64x64 => 4x4 grid of 16x16 WMMA tiles
// (16 WMMA : 16 ds_load_b128 per wave per K-step).
// A tile staged via async-to-LDS (double buffered); B tile transposed through
// a register double buffer.
// ---------------------------------------------------------------------------
template <typename OutT>
__global__ __launch_bounds__(256, 1)
void gemm_f16(const _Float16* __restrict__ A, const _Float16* __restrict__ B,
              OutT* __restrict__ C, int M, int N, int K) {
  constexpr int LDA = 40;   // f16 stride; 80B rows keep b128 ops 16B-aligned
  constexpr int BM = 128, BN = 256;
  __shared__ alignas(16) _Float16 As[2][BM * LDA];   // [m][k]
  __shared__ alignas(16) _Float16 Bs[2][BN * LDA];   // transposed: [n][k]

  const int tid  = threadIdx.x;
  const int lane = tid & 31;
  const int wave = tid >> 5;
  const int hf   = lane >> 4;
  const int ln   = lane & 15;
  const int wm   = wave & 1;    // 64-row half
  const int wn   = wave >> 1;   // 64-col quarter
  const int m0   = blockIdx.y * BM;
  const int n0   = blockIdx.x * BN;

  const int acg  = tid & 3;     // A: col group of 8 f16
  const int arow = tid >> 2;    // A: row (2 passes of 64)
  const int bng  = tid & 31;    // B: 8-col group
  const int bkr  = tid >> 5;    // B: k row (4 passes of 8)

  const int nk = K / 32;
  v8f acc[4][4] = {};
  H8  breg[4];

  auto issue_asyncA = [&](int buf, int k0) {
#pragma unroll
    for (int p = 0; p < 2; ++p) {
      int r = arow + p * 64;
      unsigned lds = (unsigned)(uintptr_t)&As[buf][r * LDA + acg * 8];
      unsigned go  = (unsigned)(((size_t)(m0 + r) * K + k0 + acg * 8) * sizeof(_Float16));
      asm volatile("global_load_async_to_lds_b128 %0, %1, %2"
                   :: "v"(lds), "v"(go), "s"(A) : "memory");
    }
  };
  auto load_Bregs = [&](int k0) {
#pragma unroll
    for (int p = 0; p < 4; ++p) {
      int kk = bkr + p * 8;
      breg[p].v = *(const v8h*)(B + (size_t)(k0 + kk) * N + n0 + bng * 8);
    }
  };
  auto store_Bregs = [&](int buf) {
#pragma unroll
    for (int p = 0; p < 4; ++p) {
      int kk = bkr + p * 8;
#pragma unroll
      for (int j = 0; j < 8; ++j) Bs[buf][(bng * 8 + j) * LDA + kk] = breg[p].e[j];
    }
  };

  // prologue: fill buffer 0
  issue_asyncA(0, 0);
  load_Bregs(0);
  store_Bregs(0);
  asm volatile("s_wait_asynccnt 0x0" ::: "memory");
  __syncthreads();

  for (int kb = 0; kb < nk; ++kb) {
    const int cur = kb & 1;
    const bool more = (kb + 1 < nk);
    if (more) {                      // overlap next-tile traffic with compute
      issue_asyncA(cur ^ 1, (kb + 1) * 32);
      load_Bregs((kb + 1) * 32);
    }

    Frag16 a[4];
#pragma unroll
    for (int i = 0; i < 4; ++i) {    // A frag: lane m=ln, K runs [8h,+8),[16+8h,+8)
      const _Float16* p = &As[cur][(wm * 64 + i * 16 + ln) * LDA];
      a[i].h[0] = *(const v8h*)(p + 8 * hf);
      a[i].h[1] = *(const v8h*)(p + 16 + 8 * hf);
    }
#pragma unroll
    for (int j = 0; j < 4; ++j) {    // B frag: lane n=ln, K runs [8h,+8),[16+8h,+8)
      Frag16 b;
      const _Float16* p = &Bs[cur][(wn * 64 + j * 16 + ln) * LDA];
      b.h[0] = *(const v8h*)(p + 8 * hf);
      b.h[1] = *(const v8h*)(p + 16 + 8 * hf);
#pragma unroll
      for (int i = 0; i < 4; ++i) acc[i][j] = wmma_f16(a[i], b, acc[i][j]);
    }

    if (more) store_Bregs(cur ^ 1);  // writes go to the other buffer: safe
    asm volatile("s_wait_asynccnt 0x0" ::: "memory");
    __syncthreads();
  }

  // C layout: lane holds col = ln, row = r + 8*hf in vgpr r
#pragma unroll
  for (int i = 0; i < 4; ++i)
#pragma unroll
    for (int j = 0; j < 4; ++j)
#pragma unroll
      for (int r = 0; r < 8; ++r) {
        int row = m0 + wm * 64 + i * 16 + r + 8 * hf;
        int col = n0 + wn * 64 + j * 16 + ln;
        C[(size_t)row * N + col] = (OutT)acc[i][j][r];
      }
}

// ---------------------------------------------------------------------------
// Neox RoPE applied in-place to q,k halves of the packed qkv (f16) buffer.
// ---------------------------------------------------------------------------
__global__ void rope_kernel(const int* __restrict__ positions, _Float16* __restrict__ qkv) {
  int idx = blockIdx.x * blockDim.x + threadIdx.x;
  const int total = BATCH * SEQ * NHEADS * (DHEAD / 2);
  if (idx >= total) return;
  int i  = idx & 63;
  int hh = (idx >> 6) & 31;
  int s  = (idx >> 11) & (SEQ - 1);
  int b  = idx >> 21;
  float pos  = (float)positions[b * SEQ + s];
  float freq = __expf(-(float)(2 * i) * (9.2103403719761836f / 128.0f));
  float ang  = pos * freq;
  float cs = __cosf(ang), sn = __sinf(ang);
  size_t base = (size_t)(b * SEQ + s) * QKV_N + hh * DHEAD;
  _Float16* q = qkv + base;
  _Float16* k = qkv + base + HDIM;
  float q1 = (float)q[i], q2 = (float)q[i + 64];
  q[i]      = (_Float16)(q1 * cs - q2 * sn);
  q[i + 64] = (_Float16)(q1 * sn + q2 * cs);
  float k1 = (float)k[i], k2 = (float)k[i + 64];
  k[i]      = (_Float16)(k1 * cs - k2 * sn);
  k[i + 64] = (_Float16)(k1 * sn + k2 * cs);
}

// ---------------------------------------------------------------------------
// Flash-style causal attention, 4 waves / block, 64-query tile, 16 rows/wave.
// Per 32-key block: 8 WMMA (Q*K^T) + 8 WMMA (P*V); all fragments preloaded
// before each WMMA chain; waves_per_eu=1 so they actually stay live.
// ---------------------------------------------------------------------------
__global__ __launch_bounds__(128, 1)
void attn_kernel(const _Float16* __restrict__ qkv, _Float16* __restrict__ ctx) {
  constexpr int ROW = QKV_N;
  constexpr int LDV = 40;
  __shared__ alignas(16) _Float16 Vt[DHEAD * LDV];   // V^T [d][k], 128x32
  __shared__ alignas(16) _Float16 Pl[4][16 * LDV];   // per-wave P 16x32

  const int tile = blockIdx.x;
  const int b    = blockIdx.y / NHEADS;
  const int h    = blockIdx.y % NHEADS;
  const int tid  = threadIdx.x;
  const int lane = tid & 31;
  const int wave = tid >> 5;
  const int hf   = lane >> 4;
  const int ln   = lane & 15;

  const size_t basebs = (size_t)b * SEQ * ROW;
  const _Float16* qp = qkv + basebs + h * DHEAD;
  const _Float16* kp = qkv + basebs + HDIM + h * DHEAD;
  const _Float16* vp = qkv + basebs + 2 * HDIM + h * DHEAD;

  const int qbase = tile * 64 + wave * 16;

  Frag16 qf[4];
  {
    const _Float16* p = qp + (size_t)(qbase + ln) * ROW;
#pragma unroll
    for (int c = 0; c < 4; ++c) {
      qf[c].h[0] = *(const v8h*)(p + c * 32 + 8 * hf);
      qf[c].h[1] = *(const v8h*)(p + c * 32 + 16 + 8 * hf);
    }
  }

  float mi[8], li[8];
  v8f oacc[8] = {};
#pragma unroll
  for (int r = 0; r < 8; ++r) { mi[r] = -INFINITY; li[r] = 0.0f; }

  const int nkb = tile * 2 + 2;
  const float scale = 0.088388347648318447f;  // 1/sqrt(128)

  for (int kb = 0; kb < nkb; ++kb) {
    const int k0 = kb * 32;

    __syncthreads();
    {  // stage V^T: 16B global loads, scalar ds-write transpose
      int kk0 = tid >> 4;
      int d0  = (tid & 15) * 8;
#pragma unroll
      for (int p = 0; p < 4; ++p) {
        int kk = kk0 + p * 8;
        H8 u;
        u.v = *(const v8h*)(vp + (size_t)(k0 + kk) * ROW + d0);
#pragma unroll
        for (int j = 0; j < 8; ++j) Vt[(d0 + j) * LDV + kk] = u.e[j];
      }
    }
    __syncthreads();

    // preload all K fragments (d-contig per key row), then run WMMA chain
    Frag16 kf[2][4];
#pragma unroll
    for (int nb = 0; nb < 2; ++nb) {
      const _Float16* kr = kp + (size_t)(k0 + nb * 16 + ln) * ROW;
#pragma unroll
      for (int c = 0; c < 4; ++c) {
        kf[nb][c].h[0] = *(const v8h*)(kr + c * 32 + 8 * hf);
        kf[nb][c].h[1] = *(const v8h*)(kr + c * 32 + 16 + 8 * hf);
      }
    }
    v8f s[2] = {};
#pragma unroll
    for (int nb = 0; nb < 2; ++nb)
#pragma unroll
      for (int c = 0; c < 4; ++c) s[nb] = wmma_f16(qf[c], kf[nb][c], s[nb]);

    // scale + causal mask + online softmax (row = r + 8*hf, col = key)
#pragma unroll
    for (int r = 0; r < 8; ++r) {
      const int qrow = qbase + r + 8 * hf;
      float s0 = s[0][r] * scale;
      float s1 = s[1][r] * scale;
      if (k0 + ln > qrow)      s0 = -INFINITY;
      if (k0 + 16 + ln > qrow) s1 = -INFINITY;
      float mx = fmaxf(s0, s1);
#pragma unroll
      for (int m = 1; m < 16; m <<= 1) mx = fmaxf(mx, __shfl_xor(mx, m, 32));
      float mnew = fmaxf(mi[r], mx);
      float corr = __expf(mi[r] - mnew);
      float p0 = __expf(s0 - mnew);
      float p1 = __expf(s1 - mnew);
      float rs = p0 + p1;
#pragma unroll
      for (int m = 1; m < 16; m <<= 1) rs += __shfl_xor(rs, m, 32);
      mi[r] = mnew;
      li[r] = li[r] * corr + rs;
#pragma unroll
      for (int cc = 0; cc < 8; ++cc) oacc[cc][r] *= corr;
      s[0][r] = p0;
      s[1][r] = p1;
    }

    // P: C-layout -> A-layout via per-wave LDS round trip
    {
      _Float16* pw = &Pl[wave][0];
#pragma unroll
      for (int r = 0; r < 8; ++r) {
        pw[(r + 8 * hf) * LDV + ln]      = (_Float16)s[0][r];
        pw[(r + 8 * hf) * LDV + 16 + ln] = (_Float16)s[1][r];
      }
    }
    asm volatile("s_wait_dscnt 0x0" ::: "memory");  // cross-lane LDS visibility
    Frag16 pf;
    {
      const _Float16* pr = &Pl[wave][ln * LDV];
      pf.h[0] = *(const v8h*)(pr + 8 * hf);
      pf.h[1] = *(const v8h*)(pr + 16 + 8 * hf);
    }

    // preload all V fragments, then O += P * V (8 d-chunks of 16)
    Frag16 vf[8];
#pragma unroll
    for (int cc = 0; cc < 8; ++cc) {
      const _Float16* vr = &Vt[(cc * 16 + ln) * LDV];
      vf[cc].h[0] = *(const v8h*)(vr + 8 * hf);
      vf[cc].h[1] = *(const v8h*)(vr + 16 + 8 * hf);
    }
#pragma unroll
    for (int cc = 0; cc < 8; ++cc) oacc[cc] = wmma_f16(pf, vf[cc], oacc[cc]);
  }

  // normalize and write f16 context [B,S,H]
#pragma unroll
  for (int cc = 0; cc < 8; ++cc)
#pragma unroll
    for (int r = 0; r < 8; ++r) {
      int q = qbase + r + 8 * hf;
      ctx[(size_t)(b * SEQ + q) * HDIM + h * DHEAD + cc * 16 + ln] =
          (_Float16)(oacc[cc][r] / li[r]);
    }
}

// ---------------------------------------------------------------------------
extern "C" void kernel_launch(void* const* d_in, const int* in_sizes, int n_in,
                              void* d_out, int out_size, void* d_ws, size_t ws_size,
                              hipStream_t stream) {
  const int*   positions = (const int*)d_in[0];
  const float* hidden    = (const float*)d_in[1];
  const int*   qkv_qw    = (const int*)d_in[2];
  const int*   qkv_qz    = (const int*)d_in[3];
  const float* qkv_sc    = (const float*)d_in[4];
  const int*   o_qw      = (const int*)d_in[5];
  const int*   o_qz      = (const int*)d_in[6];
  const float* o_sc      = (const float*)d_in[7];
  (void)in_sizes; (void)n_in; (void)out_size; (void)ws_size;

  // workspace carve-up (all segments 16B-aligned): ~208 MB
  char* ws = (char*)d_ws;
  _Float16* Wqkv  = (_Float16*)ws; ws += (size_t)HDIM * QKV_N * sizeof(_Float16);
  _Float16* Wo    = (_Float16*)ws; ws += (size_t)HDIM * HDIM * sizeof(_Float16);
  _Float16* hid16 = (_Float16*)ws; ws += (size_t)BATCH * SEQ * HDIM * sizeof(_Float16);
  _Float16* qkvb  = (_Float16*)ws; ws += (size_t)BATCH * SEQ * QKV_N * sizeof(_Float16);
  _Float16* ctxb  = (_Float16*)ws; ws += (size_t)BATCH * SEQ * HDIM * sizeof(_Float16);

  const int M = BATCH * SEQ;

  { int n = HDIM * (QKV_N / 8);
    dequant_awq<<<(n + 255) / 256, 256, 0, stream>>>(qkv_qw, qkv_qz, qkv_sc, Wqkv, HDIM, QKV_N); }
  { int n = HDIM * (HDIM / 8);
    dequant_awq<<<(n + 255) / 256, 256, 0, stream>>>(o_qw, o_qz, o_sc, Wo, HDIM, HDIM); }
  { int n = M * HDIM;
    f32_to_f16<<<(n + 255) / 256, 256, 0, stream>>>(hidden, hid16, n); }

  gemm_f16<_Float16><<<dim3(QKV_N / 256, M / 128), 256, 0, stream>>>(
      hid16, Wqkv, qkvb, M, QKV_N, HDIM);

  { int n = BATCH * SEQ * NHEADS * (DHEAD / 2);
    rope_kernel<<<(n + 255) / 256, 256, 0, stream>>>(positions, qkvb); }

  attn_kernel<<<dim3(SEQ / 64, BATCH * NHEADS), 128, 0, stream>>>(qkvb, ctxb);

  gemm_f16<float><<<dim3(HDIM / 256, M / 128), 256, 0, stream>>>(
      ctxb, Wo, (float*)d_out, M, HDIM, HDIM);
}
